// SingleSAEv5_12841952215220
// MI455X (gfx1250) — matrix operand order repaired and checked by reading, not compiled, not run on hardware
//
#include <hip/hip_runtime.h>
#include <hip/hip_bf16.h>

// ---------------------------------------------------------------------------
// Top-k sparse autoencoder forward for MI455X (gfx1250, wave32).
//  N=8192, d=768, F=32768, k=32.
//  - encoder GEMM on v_wmma_f32_16x16x32_bf16 (compute-dominant: 412 GFLOP)
//  - top-k fused into the GEMM kernel => h_pre (1 GB) never touches HBM
//  - enc_w/dec_wT as bf16 (48 MB each) => resident in 192 MB L2
//  - F split x2 per row-block (256 WGs) + tiny top-k merge kernel
// ---------------------------------------------------------------------------

typedef __bf16 bf16_t;
typedef __attribute__((ext_vector_type(16))) __bf16 v16bf;
typedef __attribute__((ext_vector_type(8)))  __bf16 v8bf;
typedef __attribute__((ext_vector_type(4)))  __bf16 v4bf;
typedef __attribute__((ext_vector_type(8)))  float  v8f;
typedef __attribute__((ext_vector_type(4)))  float  v4f;

#define NROWS 8192
#define DDIM  768
#define FDIM  32768
#define KTOP  32

#define MTILE 64          // rows per block
#define NTILE 256         // cols per F-chunk (8 waves x 2 x 16)
#define FSPLIT 2          // F halves scanned by independent blocks
#define FHALF (FDIM / FSPLIT)
#define AS_STRIDE 776     // bf16 elems per LDS A row (768 + pad, 16B-aligned steps)
#define HS_STRIDE 68      // f32 per LDS H column (64 rows + pad)
#define LDS_BYTES (MTILE*AS_STRIDE*2 + NTILE*HS_STRIDE*4)   // 99328 + 69632 = 168960

#define NEG_INF (-__builtin_inff())

__device__ __forceinline__ int lane_id() {
  return __builtin_amdgcn_mbcnt_hi(~0u, __builtin_amdgcn_mbcnt_lo(~0u, 0u));
}
__device__ __forceinline__ float bperm_f(float v, int srcLane) {
  int i = __builtin_amdgcn_ds_bpermute(srcLane << 2, __builtin_bit_cast(int, v));
  return __builtin_bit_cast(float, i);
}
__device__ __forceinline__ int bperm_i(int v, int srcLane) {
  return __builtin_amdgcn_ds_bpermute(srcLane << 2, v);
}

// ---------------------------------------------------------------------------
// 1) LayerNorm: xn = (x - mean)/std - pre_bias  (store bf16), keep mean/std.
// ---------------------------------------------------------------------------
__global__ __launch_bounds__(256)
void ln_prep_kernel(const float* __restrict__ x,
                    const float* __restrict__ pre_bias,
                    bf16_t* __restrict__ xn,
                    float* __restrict__ meanArr,
                    float* __restrict__ stdArr) {
  __shared__ float sS[256];
  __shared__ float sS2[256];
  const int row = blockIdx.x;
  const int tid = threadIdx.x;
  const float* xr = x + (size_t)row * DDIM;
  float v0 = xr[tid], v1 = xr[tid + 256], v2 = xr[tid + 512];
  sS[tid]  = v0 + v1 + v2;
  sS2[tid] = v0 * v0 + v1 * v1 + v2 * v2;
  __syncthreads();
  for (int st = 128; st > 0; st >>= 1) {
    if (tid < st) { sS[tid] += sS[tid + st]; sS2[tid] += sS2[tid + st]; }
    __syncthreads();
  }
  const float mean = sS[0] * (1.0f / DDIM);
  const float var  = sS2[0] * (1.0f / DDIM) - mean * mean;
  const float stdv = sqrtf(var + 1e-5f);
  const float inv  = 1.0f / stdv;
  if (tid == 0) { meanArr[row] = mean; stdArr[row] = stdv; }
  bf16_t* xo = xn + (size_t)row * DDIM;
  xo[tid]       = (bf16_t)((v0 - mean) * inv - pre_bias[tid]);
  xo[tid + 256] = (bf16_t)((v1 - mean) * inv - pre_bias[tid + 256]);
  xo[tid + 512] = (bf16_t)((v2 - mean) * inv - pre_bias[tid + 512]);
}

// ---------------------------------------------------------------------------
// 2) fp32 -> bf16 bulk convert (enc_w), vectorized b128 in / b64 out.
// ---------------------------------------------------------------------------
__global__ __launch_bounds__(256)
void cvt_bf16_kernel(const float4* __restrict__ src, v4bf* __restrict__ dst, long n4) {
  long i = (long)blockIdx.x * blockDim.x + threadIdx.x;
  const long stride = (long)gridDim.x * blockDim.x;
  for (; i < n4; i += stride) {
    float4 f = src[i];
    v4bf o;
    o[0] = (bf16_t)f.x; o[1] = (bf16_t)f.y; o[2] = (bf16_t)f.z; o[3] = (bf16_t)f.w;
    dst[i] = o;
  }
}

// ---------------------------------------------------------------------------
// 3) dec_w (d x F, f32) -> dec_wT (F x d, bf16) tiled transpose via LDS.
// ---------------------------------------------------------------------------
__global__ __launch_bounds__(256)
void transpose_dec_kernel(const float* __restrict__ dec_w, bf16_t* __restrict__ decT) {
  __shared__ float tile[32][33];
  const int fBase = blockIdx.x * 32;
  const int dBase = blockIdx.y * 32;
  const int tx = threadIdx.x, ty = threadIdx.y;  // 32 x 8
#pragma unroll
  for (int q = 0; q < 4; q++) {
    tile[ty + q * 8][tx] = dec_w[(size_t)(dBase + ty + q * 8) * FDIM + fBase + tx];
  }
  __syncthreads();
#pragma unroll
  for (int q = 0; q < 4; q++) {
    decT[(size_t)(fBase + ty + q * 8) * DDIM + dBase + tx] = (bf16_t)tile[tx][ty + q * 8];
  }
}

// ---------------------------------------------------------------------------
// 4) Fused encoder GEMM (WMMA bf16) + streaming exact top-32 per row.
//    256 threads (8 waves), 64 rows, scans one F-half in 256-col chunks.
//    Per k-step: 4 A-fragments (LDS) x 2 B-fragments (global/L2) -> 8 WMMAs,
//    all fragment loads issued before the WMMA group (one wait covers all).
//    Top-k buffer: 32 entries/row == one entry per wave32 lane (registers).
// ---------------------------------------------------------------------------
__global__ __launch_bounds__(256)
void enc_topk_kernel(const bf16_t* __restrict__ xn,
                     const bf16_t* __restrict__ encw,
                     const float* __restrict__ latent_bias,
                     float* __restrict__ pv,      // partial top-k vals [row][2*KTOP]
                     int* __restrict__ pi) {      // partial top-k idx  [row][2*KTOP]
  extern __shared__ __align__(16) char lds_raw[];
  bf16_t* As = (bf16_t*)lds_raw;                               // [64][776] bf16
  float*  Hs = (float*)(lds_raw + MTILE * AS_STRIDE * 2);      // [256][68] f32, col-major

  const int tid  = threadIdx.x;
  const int wave = tid >> 5;
  const int lane = lane_id();
  const int half = lane >> 4;
  const int l16  = lane & 15;
  const int rbase = (blockIdx.x >> 1) * MTILE;
  const int fhalf = blockIdx.x & 1;
  const int f0    = fhalf * FHALF;

  // ---- stage the 64 x 768 bf16 A tile into LDS (16B chunks) ----
  {
    const int cpr = DDIM / 8;  // 96 chunks per row
    for (int c = tid; c < MTILE * cpr; c += 256) {
      const int r = c / cpr, k8 = c % cpr;
      v8bf v = *(const v8bf*)(xn + (size_t)(rbase + r) * DDIM + k8 * 8);
      *(v8bf*)(As + r * AS_STRIDE + k8 * 8) = v;
    }
  }
  __syncthreads();

  // ---- per-wave top-k state: 8 rows/wave, one (val,idx) slot per lane ----
  float bufv[8]; int bufi[8]; float thr[8];
#pragma unroll
  for (int r = 0; r < 8; r++) { bufv[r] = NEG_INF; bufi[r] = 0; thr[r] = NEG_INF; }

  const int colloc = wave * 16 + l16;  // local column (group 0) this lane produces

  for (int fbase = f0; fbase < f0 + FHALF; fbase += NTILE) {
    v8f acc[2][4] = {};  // 2 col-groups x 4 row-tiles
    const bf16_t* wrow0 = encw + (size_t)(fbase + colloc) * DDIM;
    const bf16_t* wrow1 = encw + (size_t)(fbase + 128 + colloc) * DDIM;

    // prefetch next chunk's B rows into L2/L0 while this chunk computes
    if (fbase + NTILE < f0 + FHALF) {
      __builtin_prefetch(wrow0 + (size_t)NTILE * DDIM, 0, 1);
      __builtin_prefetch(wrow1 + (size_t)NTILE * DDIM, 0, 1);
    }

#pragma unroll 4
    for (int kb = 0; kb < DDIM; kb += 32) {
      // A fragments for the 4 row-tiles: two 16B K-slices each (ISA layout)
      v16bf a[4];
#pragma unroll
      for (int t = 0; t < 4; t++) {
        const bf16_t* arow = As + (t * 16 + l16) * AS_STRIDE + kb + 8 * half;
        v8bf a0 = *(const v8bf*)(arow);
        v8bf a1 = *(const v8bf*)(arow + 16);
        a[t] = __builtin_shufflevector(a0, a1, 0,1,2,3,4,5,6,7,8,9,10,11,12,13,14,15);
      }
      // B fragments: two 16B K-slices from each lane's enc_w row (ISA layout, matches A)
      v8bf b0l = *(const v8bf*)(wrow0 + kb + 8 * half);
      v8bf b0h = *(const v8bf*)(wrow0 + kb + 8 * half + 16);
      v16bf b0 = __builtin_shufflevector(b0l, b0h, 0,1,2,3,4,5,6,7,8,9,10,11,12,13,14,15);
      v8bf b1l = *(const v8bf*)(wrow1 + kb + 8 * half);
      v8bf b1h = *(const v8bf*)(wrow1 + kb + 8 * half + 16);
      v16bf b1 = __builtin_shufflevector(b1l, b1h, 0,1,2,3,4,5,6,7,8,9,10,11,12,13,14,15);
#pragma unroll
      for (int t = 0; t < 4; t++) {
        acc[0][t] = __builtin_amdgcn_wmma_f32_16x16x32_bf16(
            false, a[t], false, b0, (short)0, acc[0][t], false, false);
        acc[1][t] = __builtin_amdgcn_wmma_f32_16x16x32_bf16(
            false, a[t], false, b1, (short)0, acc[1][t], false, false);
      }
    }

    // bias + tile store (C elem r -> row t*16 + 8*half + r, col fixed per lane)
#pragma unroll
    for (int g = 0; g < 2; g++) {
      const float lb = latent_bias[fbase + g * 128 + colloc];
#pragma unroll
      for (int t = 0; t < 4; t++) {
        float* hp = Hs + (g * 128 + colloc) * HS_STRIDE + t * 16 + 8 * half;
        v4f lo = {acc[g][t][0] + lb, acc[g][t][1] + lb, acc[g][t][2] + lb, acc[g][t][3] + lb};
        v4f hi = {acc[g][t][4] + lb, acc[g][t][5] + lb, acc[g][t][6] + lb, acc[g][t][7] + lb};
        *(v4f*)(hp)     = lo;   // 8 consecutive rows of this column -> 2x b128 ds stores
        *(v4f*)(hp + 4) = hi;
      }
    }
    __syncthreads();

    // ---- streaming top-k update: wave owns local rows [wave*8, wave*8+8) ----
#pragma unroll 1
    for (int r = 0; r < 8; r++) {
      const int rloc = wave * 8 + r;
      float cand[8];
#pragma unroll
      for (int q = 0; q < 8; q++) cand[q] = Hs[(lane * 8 + q) * HS_STRIDE + rloc];
      float t0 = thr[r];
      while (true) {
        // per-lane best of 8, then wave argmax (butterfly via ds_bpermute)
        float bv = cand[0]; int bc = lane * 8;
#pragma unroll
        for (int q = 1; q < 8; q++)
          if (cand[q] > bv) { bv = cand[q]; bc = lane * 8 + q; }
        for (int off = 16; off > 0; off >>= 1) {
          float ov = bperm_f(bv, lane ^ off);
          int   oc = bperm_i(bc, lane ^ off);
          if (ov > bv || (ov == bv && oc < bc)) { bv = ov; bc = oc; }
        }
        if (!(bv > t0)) break;          // common case: nothing beats current kth-max
        // replace buffer min (wave argmin over the 32 per-lane slots)
        float mv = bufv[r]; int ml = lane;
        for (int off = 16; off > 0; off >>= 1) {
          float ov = bperm_f(mv, lane ^ off);
          int   ol = bperm_i(ml, lane ^ off);
          if (ov < mv || (ov == mv && ol < ml)) { mv = ov; ml = ol; }
        }
        if (lane == ml) { bufv[r] = bv; bufi[r] = fbase + bc; }
        if ((bc >> 3) == lane) cand[bc & 7] = NEG_INF;   // consume candidate
        float nm = bufv[r];                              // new threshold = min(buf)
        for (int off = 16; off > 0; off >>= 1) nm = fminf(nm, bperm_f(nm, lane ^ off));
        t0 = nm;
      }
      thr[r] = t0;
    }
    __syncthreads();
  }

  // ---- emit this half's top-32 per row (raw; ReLU applied in merge) ----
#pragma unroll
  for (int r = 0; r < 8; r++) {
    const int grow = rbase + wave * 8 + r;
    pv[grow * (2 * KTOP) + fhalf * KTOP + lane] = bufv[r];
    pi[grow * (2 * KTOP) + fhalf * KTOP + lane] = bufi[r];
  }
}

// ---------------------------------------------------------------------------
// 5) Merge two partial top-32 sets -> global top-32 (one wave per row).
//    Each lane holds 2 candidates; 32 argmax-extract rounds.
// ---------------------------------------------------------------------------
__global__ __launch_bounds__(256)
void topk_merge_kernel(const float* __restrict__ pv, const int* __restrict__ pi,
                       float* __restrict__ topv, int* __restrict__ topi) {
  const int wave = threadIdx.x >> 5;
  const int lane = lane_id();
  const int row = blockIdx.x * 8 + wave;
  float c0 = pv[row * 64 + lane];
  float c1 = pv[row * 64 + 32 + lane];
  const int i0 = pi[row * 64 + lane];
  const int i1 = pi[row * 64 + 32 + lane];
#pragma unroll 1
  for (int s = 0; s < KTOP; s++) {
    float bv; int bc;
    if (c0 >= c1) { bv = c0; bc = lane * 2; } else { bv = c1; bc = lane * 2 + 1; }
    for (int off = 16; off > 0; off >>= 1) {
      float ov = bperm_f(bv, lane ^ off);
      int   oc = bperm_i(bc, lane ^ off);
      if (ov > bv || (ov == bv && oc < bc)) { bv = ov; bc = oc; }
    }
    const int src = bc >> 1;                       // uniform
    const int idx = bperm_i((bc & 1) ? i1 : i0, src);
    if (lane == 0) {
      topv[row * KTOP + s] = fmaxf(bv, 0.0f);      // ReLU of kept value
      topi[row * KTOP + s] = idx;
    }
    if (lane == src) { if (bc & 1) c1 = NEG_INF; else c0 = NEG_INF; }
  }
}

// ---------------------------------------------------------------------------
// 6) Sparse decode + de-normalize:
//    x_hat = (sum_k v_k * decT[idx_k,:] + pre_bias) * std + mean
//    192 threads x 4 contiguous elems -> b64 bf16 loads from L2-resident decT.
// ---------------------------------------------------------------------------
__global__ __launch_bounds__(192)
void decode_kernel(const float* __restrict__ topv, const int* __restrict__ topi,
                   const bf16_t* __restrict__ decT,
                   const float* __restrict__ pre_bias,
                   const float* __restrict__ meanArr, const float* __restrict__ stdArr,
                   float* __restrict__ out) {
  const int row = blockIdx.x;
  const int j0 = threadIdx.x * 4;
  const float mean = meanArr[row];
  const float stdv = stdArr[row];
  float4 pb = *(const float4*)(pre_bias + j0);
  float a0 = pb.x, a1 = pb.y, a2 = pb.z, a3 = pb.w;
#pragma unroll 4
  for (int t = 0; t < KTOP; t++) {
    const float v = topv[row * KTOP + t];        // uniform -> scalarized
    const int idx = topi[row * KTOP + t];
    v4bf w = *(const v4bf*)(decT + (size_t)idx * DDIM + j0);
    a0 += v * (float)w[0];
    a1 += v * (float)w[1];
    a2 += v * (float)w[2];
    a3 += v * (float)w[3];
  }
  float4 o = {a0 * stdv + mean, a1 * stdv + mean, a2 * stdv + mean, a3 * stdv + mean};
  *(float4*)(out + (size_t)row * DDIM + j0) = o;
}

// ---------------------------------------------------------------------------
// Launch. Workspace layout (needs ~119.7 MB):
//   [0)            xn bf16            12,582,912 B
//   [12,582,912)   mean f32               32,768 B
//   [12,615,680)   std  f32               32,768 B
//   [12,648,448)   enc_w bf16         50,331,648 B
//   [62,980,096)   dec_wT bf16        50,331,648 B
//   [113,311,744)  partial topv f32    2,097,152 B
//   [115,408,896)  partial topi i32    2,097,152 B
//   [117,506,048)  topv f32            1,048,576 B
//   [118,554,624)  topi i32            1,048,576 B
// ---------------------------------------------------------------------------
extern "C" void kernel_launch(void* const* d_in, const int* in_sizes, int n_in,
                              void* d_out, int out_size, void* d_ws, size_t ws_size,
                              hipStream_t stream) {
  (void)in_sizes; (void)n_in; (void)out_size; (void)ws_size;
  const float* x           = (const float*)d_in[0];
  const float* pre_bias    = (const float*)d_in[1];
  const float* latent_bias = (const float*)d_in[2];
  const float* enc_w       = (const float*)d_in[3];
  const float* dec_w       = (const float*)d_in[4];
  // d_in[5] is k (==32, baked in as KTOP)

  char* ws = (char*)d_ws;
  bf16_t* xn      = (bf16_t*)(ws);
  float*  meanArr = (float*)(ws + 12582912);
  float*  stdArr  = (float*)(ws + 12615680);
  bf16_t* encb    = (bf16_t*)(ws + 12648448);
  bf16_t* decT    = (bf16_t*)(ws + 62980096);
  float*  pvP     = (float*)(ws + 113311744);
  int*    piP     = (int*)(ws + 115408896);
  float*  topv    = (float*)(ws + 117506048);
  int*    topi    = (int*)(ws + 118554624);

  ln_prep_kernel<<<NROWS, 256, 0, stream>>>(x, pre_bias, xn, meanArr, stdArr);
  cvt_bf16_kernel<<<4096, 256, 0, stream>>>((const float4*)enc_w, (v4bf*)encb,
                                            (long)FDIM * DDIM / 4);
  transpose_dec_kernel<<<dim3(FDIM / 32, DDIM / 32, 1), dim3(32, 8, 1), 0, stream>>>(dec_w, decT);
  enc_topk_kernel<<<(NROWS / MTILE) * FSPLIT, 256, LDS_BYTES, stream>>>(xn, encb, latent_bias,
                                                                        pvP, piP);
  topk_merge_kernel<<<NROWS / 8, 256, 0, stream>>>(pvP, piP, topv, topi);
  decode_kernel<<<NROWS, 192, 0, stream>>>(topv, topi, decT, pre_bias, meanArr, stdArr,
                                           (float*)d_out);
}
